// FeatureInteractionModel_71829033059095
// MI455X (gfx1250) — matrix operand order repaired and checked by reading, not compiled, not run on hardware
//
#include <hip/hip_runtime.h>

// ---------------------------------------------------------------------------
// y[i] = beta0 + X[i,:].beta + X[i,:] @ Theta @ X[i,:]^T  (Theta strictly upper)
// N=16384 rows, P=1024 features.
// Split-bf16 (hi+lo) WMMA GEMM, fp32 accumulate (hi*hi + hi*lo + lo*hi).
// Theta pre-packed B-tile-ready; B tiles double-buffered in LDS via
// GLOBAL_LOAD_ASYNC_TO_LDS_B128 (ASYNCcnt); A operands from LDS; X read from
// HBM exactly once per 32-row block.
// ---------------------------------------------------------------------------

typedef __bf16 bf16;
typedef __attribute__((ext_vector_type(16))) __bf16 v16bf;
typedef __attribute__((ext_vector_type(8)))  __bf16 v8bf;
typedef __attribute__((ext_vector_type(8)))  float  v8f;
typedef __attribute__((ext_vector_type(4)))  float  v4f;

#define PDIM   1024
#define NROWS  16384
#define MTILE  32            // rows of X per workgroup
#define THREADS 128          // 4 waves (wave32)
#define LDS_PITCH (PDIM + 8) // halfs; pad so 16B chunks spread across banks

// ---------------------------------------------------------------------------
// Kernel 1: scatter packed strictly-upper theta into dense bf16 hi/lo planes,
// stored B-tile-ready: element (j, c) -> [(c/16)*PDIM*16 + (j/32)*512 + (c%16)*32 + (j%32)].
// A WMMA B-tile (32 k-rows x 16 cols) is then 1KB contiguous.
// ---------------------------------------------------------------------------
__global__ __launch_bounds__(256)
void theta_scatter_kernel(const float* __restrict__ theta_upper,
                          bf16* __restrict__ tHi, bf16* __restrict__ tLo) {
  int idx = blockIdx.x * 256 + threadIdx.x;     // one thread per (j, c)
  int j = idx >> 10;
  int c = idx & (PDIM - 1);
  float v = 0.0f;
  if (c > j) {
    int pk = j * PDIM - ((j * (j + 1)) >> 1) + (c - j - 1);
    v = theta_upper[pk];
  }
  bf16 h = (bf16)v;
  bf16 l = (bf16)(v - (float)h);
  int dst = ((c >> 4) << 14) + ((j >> 5) << 9) + ((c & 15) << 5) + (j & 31);
  tHi[dst] = h;
  tLo[dst] = l;
}

// ---------------------------------------------------------------------------
// Async-stage one B tile pair (hi+lo, 32 k-rows x 16 cols = 2KB) into LDS.
// Four GLOBAL_LOAD_ASYNC_TO_LDS_B128 ops (each: 32 lanes x 16B = 512B).
// Tile linear order in LDS == tile linear order in the packed Theta planes.
// ---------------------------------------------------------------------------
__device__ __forceinline__ void stage_tile(const bf16* tHi, const bf16* tLo,
                                           unsigned ldsBase, int lane,
                                           int ct, int kb) {
  const size_t goff = ((size_t)ct << 14) + (size_t)kb * 16 + (size_t)lane * 8;
  unsigned long long gh0 = (unsigned long long)(const void*)(tHi + goff);
  unsigned long long gh1 = (unsigned long long)(const void*)(tHi + goff + 256);
  unsigned long long gl0 = (unsigned long long)(const void*)(tLo + goff);
  unsigned long long gl1 = (unsigned long long)(const void*)(tLo + goff + 256);
  unsigned la = ldsBase + (unsigned)lane * 16;
  asm volatile(
      "global_load_async_to_lds_b128 %0, %4, off\n\t"
      "global_load_async_to_lds_b128 %1, %5, off\n\t"
      "global_load_async_to_lds_b128 %2, %6, off\n\t"
      "global_load_async_to_lds_b128 %3, %7, off"
      :: "v"(la), "v"(la + 512u), "v"(la + 1024u), "v"(la + 1536u),
         "v"(gh0), "v"(gh1), "v"(gl0), "v"(gl1)
      : "memory");
}

// ---------------------------------------------------------------------------
// Kernel 2: fused  (X@Theta) row-dot X  +  X@beta  +  beta0.
// Workgroup: 4 waves, 32 rows. Wave w: rows 16*(w&1).. , c-tile group w>>1.
// ---------------------------------------------------------------------------
__global__ __launch_bounds__(THREADS)
void quad_form_kernel(const float* __restrict__ X,
                      const float* __restrict__ beta0,
                      const float* __restrict__ beta,
                      const bf16* __restrict__ tHi,
                      const bf16* __restrict__ tLo,
                      float* __restrict__ out) {
  __shared__ bf16  Xh[MTILE][LDS_PITCH];        // hi part of X rows
  __shared__ bf16  Xl[MTILE][LDS_PITCH];        // lo residual of X rows
  __shared__ float ylin[MTILE][4];              // partial X@beta per quarter
  __shared__ float yquad[4][16];                // per-wave quadratic partials
  // B staging: [wave][buf][plane hi/lo][512 halfs] -> 16KB, per-wave private
  __shared__ __align__(16) bf16 Bst[4][2][2][512];

  const int t       = threadIdx.x;
  const int wave    = t >> 5;
  const int lane    = t & 31;
  const int rowBase = blockIdx.x * MTILE;

  // ---- phase 1: stage X block -> LDS split bf16; fold in linear term ------
  {
    const int rr = t >> 2;        // 0..31 : row within tile
    const int q  = t & 3;         // 0..3  : quarter of the row (256 cols)
    const float* xr = X + (size_t)(rowBase + rr) * PDIM + q * (PDIM / 4);
    const float* br = beta + q * (PDIM / 4);
    float acc = 0.0f;
    for (int i = 0; i < PDIM / 4; i += 4) {
      v4f x = *(const v4f*)(xr + i);
      v4f b = *(const v4f*)(br + i);
#pragma unroll
      for (int e = 0; e < 4; ++e) {
        float xv = x[e];
        bf16 h = (bf16)xv;
        bf16 l = (bf16)(xv - (float)h);
        Xh[rr][q * (PDIM / 4) + i + e] = h;
        Xl[rr][q * (PDIM / 4) + i + e] = l;
        acc += xv * b[e];
      }
    }
    ylin[rr][q] = acc;
  }
  __syncthreads();

  // ---- phase 2: WMMA over c-tiles (16 cols) x k-tiles (32 deep) -----------
  const int m0    = (wave & 1) * 16;   // row sub-tile of this wave
  const int cg    = wave >> 1;         // c-tile parity handled by this wave
  const int half  = lane >> 4;         // half-wave (ISA A/C layout uses it)
  const int mlane = lane & 15;

  const unsigned ldsB0 = (unsigned)(unsigned long long)&Bst[wave][0][0][0];
  const unsigned ldsB1 = ldsB0 + 2048u;

  float yq[8] = {0.f, 0.f, 0.f, 0.f, 0.f, 0.f, 0.f, 0.f};

  // prologue: stage first tile into buffer 0
  stage_tile(tHi, tLo, ldsB0, lane, cg, 0);

  int step = 0;
  for (int ct = cg; ct < PDIM / 16; ct += 2) {
    v8f z = {0.f, 0.f, 0.f, 0.f, 0.f, 0.f, 0.f, 0.f};

    for (int kb = 0; kb < PDIM; kb += 32, ++step) {
      const int buf = step & 1;

      // branchless double-buffer: always stage a "next" tile; past the end of
      // the schedule it wraps to (cg,0) as a harmless dummy so the steady
      // state has no stage-vs-drain branch. Async loads complete in order,
      // so waiting ASYNCcnt<=4 guarantees the CURRENT buffer is resident.
      int nct = ct, nkb = kb + 32;
      if (nkb == PDIM) { nkb = 0; nct = ct + 2; }
      if (nct >= PDIM / 16) { nct = cg; nkb = 0; }   // dummy wrap (never read)
      stage_tile(tHi, tLo, buf ? ldsB0 : ldsB1, lane, nct, nkb);
      asm volatile("s_wait_asynccnt 0x4" ::: "memory");

      // A operand (16x32 bf16): lane l = row m0+(l&15);
      // lanes 0-15 hold K kb+0..7 / kb+16..23, lanes 16-31 kb+8..15 / 24..31
      const int arow = m0 + mlane;
      const int k0   = kb + 8 * half;
      union { v16bf v; v8bf p[2]; } aH, aL, bH, bL;
      aH.p[0] = *(const v8bf*)&Xh[arow][k0];
      aH.p[1] = *(const v8bf*)&Xh[arow][k0 + 16];
      aL.p[0] = *(const v8bf*)&Xl[arow][k0];
      aL.p[1] = *(const v8bf*)&Xl[arow][k0 + 16];

      // B operand from the staged LDS tile: lane l = col l&15, K as for A
      bH.p[0] = *(const v8bf*)&Bst[wave][buf][0][mlane * 32 + 8 * half];
      bH.p[1] = *(const v8bf*)&Bst[wave][buf][0][mlane * 32 + 16 + 8 * half];
      bL.p[0] = *(const v8bf*)&Bst[wave][buf][1][mlane * 32 + 8 * half];
      bL.p[1] = *(const v8bf*)&Bst[wave][buf][1][mlane * 32 + 16 + 8 * half];

      // fp32-accurate product via 3 bf16 WMMAs: hi*hi + hi*lo + lo*hi
      z = __builtin_amdgcn_wmma_f32_16x16x32_bf16(false, aH.v, false, bH.v,
                                                  (short)0, z, false, false);
      z = __builtin_amdgcn_wmma_f32_16x16x32_bf16(false, aH.v, false, bL.v,
                                                  (short)0, z, false, false);
      z = __builtin_amdgcn_wmma_f32_16x16x32_bf16(false, aL.v, false, bH.v,
                                                  (short)0, z, false, false);
    }

    // fold: y[m] += sum_n Z[m,n] * X[m, ct*16+n]
    // C layout: VGPR r, lane l -> (m = r + 8*(l>>4), n = l&15)
    const int c = ct * 16 + mlane;
#pragma unroll
    for (int r = 0; r < 8; ++r) {
      const int row = m0 + r + 8 * half;
      float xv = (float)Xh[row][c] + (float)Xl[row][c];
      yq[r] += z[r] * xv;
    }
  }

  // ---- phase 3: reduce the 16 n-lanes sharing each row --------------------
#pragma unroll
  for (int r = 0; r < 8; ++r) {
    float v = yq[r];
    v += __shfl_xor(v, 1, 32);
    v += __shfl_xor(v, 2, 32);
    v += __shfl_xor(v, 4, 32);
    v += __shfl_xor(v, 8, 32);
    yq[r] = v;
  }
  if (mlane == 0) {
#pragma unroll
    for (int r = 0; r < 8; ++r)
      yquad[wave][r + 8 * half] = yq[r];
  }
  __syncthreads();

  // combine the two c-split waves per row group; add linear term + beta0
  if (t < MTILE) {
    const int g  = t >> 4;      // row group 0/1 -> waves g and g+2
    const int rl = t & 15;
    float quad = yquad[g][rl] + yquad[g + 2][rl];
    float lin  = ylin[t][0] + ylin[t][1] + ylin[t][2] + ylin[t][3];
    out[rowBase + t] = beta0[0] + lin + quad;
  }
}

// ---------------------------------------------------------------------------
extern "C" void kernel_launch(void* const* d_in, const int* in_sizes, int n_in,
                              void* d_out, int out_size, void* d_ws, size_t ws_size,
                              hipStream_t stream) {
  const float* X           = (const float*)d_in[0];
  const float* beta0       = (const float*)d_in[1];
  const float* beta        = (const float*)d_in[2];
  const float* theta_upper = (const float*)d_in[3];

  bf16* tHi = (bf16*)d_ws;                       // 2 MB
  bf16* tLo = tHi + (size_t)PDIM * PDIM;         // 2 MB
  float* out = (float*)d_out;

  theta_scatter_kernel<<<(PDIM * PDIM) / 256, 256, 0, stream>>>(theta_upper, tHi, tLo);
  quad_form_kernel<<<NROWS / MTILE, THREADS, 0, stream>>>(X, beta0, beta, tHi, tLo, out);
}
